// MultiHeadAttention_8976481648932
// MI455X (gfx1250) — matrix operand order repaired and checked by reading, not compiled, not run on hardware
//
#include <hip/hip_runtime.h>

// ---------------- CDNA5 WMMA types ----------------
typedef __attribute__((ext_vector_type(16))) __bf16 v16bf;
typedef __attribute__((ext_vector_type(8)))  float  v8f;
typedef __attribute__((ext_vector_type(4)))  unsigned int v4u;
typedef __attribute__((ext_vector_type(8)))  unsigned int v8u;

// Problem constants
constexpr int B_ = 4, T_ = 2048, C_ = 1024, H_ = 16, D_ = 64;
constexpr int BT_ = B_ * T_;        // 8192
constexpr int C3_ = 3 * C_;         // 3072

// ---------------- CDNA5 async global->LDS copy (ASYNCcnt path) ----------------
__device__ __forceinline__ void async_copy_b128(void* lds_dst, const void* gsrc) {
    unsigned lds_off = (unsigned)(unsigned long long)lds_dst;   // AS(3) byte offset
    asm volatile("global_load_async_to_lds_b128 %0, %1, off"
                 :
                 : "v"(lds_off), "v"((unsigned long long)gsrc)
                 : "memory");
}
__device__ __forceinline__ void wait_async0() {
    asm volatile("s_wait_asynccnt 0" ::: "memory");
}

// ---------------- CDNA5 Tensor Data Mover (TENSORcnt path) ----------------
// One wave issues a whole 2D bf16 tile (global -> LDS) via a D# descriptor.
// D# group0: [1:0]=count(1), [63:32]=lds_addr, [120:64]=global_addr, [127:126]=type(2)
// D# group1: [17:16]=data_size(1 => 2B), tensor_dim0/1, tile_dim0/1, dim0_stride
__device__ __forceinline__ void tdm_load_2d(unsigned lds_off, const void* gaddr,
                                            unsigned tensor_d0, unsigned tensor_d1,
                                            unsigned tile_d0, unsigned tile_d1,
                                            unsigned stride0 /* elements */) {
    unsigned long long ga = (unsigned long long)gaddr;
    v4u g0 = { 1u,                                   // count = 1 valid descriptor
               lds_off,                              // LDS byte address
               (unsigned)ga,                         // global addr [31:0]
               (unsigned)((ga >> 32) & 0x1FFFFFFu)   // global addr [56:32]
                   | (2u << 30) };                   // type = 2 ("image")
    v8u g1 = { 1u << 16,                                         // data_size=1 (2B)
               (tensor_d0 & 0xFFFFu) << 16,                      // tdim0[15:0]
               (tensor_d0 >> 16) | ((tensor_d1 & 0xFFFFu) << 16),// tdim0[31:16]|tdim1[15:0]
               (tensor_d1 >> 16) | (tile_d0 << 16),              // tdim1[31:16]|tile0
               tile_d1,                                          // tile1 (tile2=0)
               stride0,                                          // dim0_stride[31:0]
               0u, 0u };
    asm volatile("tensor_load_to_lds %0, %1"
                 :
                 : "s"(g0), "s"(g1)
                 : "memory");
}
__device__ __forceinline__ void wait_tensor0() {
    __builtin_amdgcn_s_wait_tensorcnt(0);
}
__device__ __forceinline__ unsigned lds_off_of(const void* p) {
    return (unsigned)(unsigned long long)p;
}

// =======================================================================
// fp32 -> bf16 elementwise convert
// =======================================================================
__global__ __launch_bounds__(256) void f2bf_kernel(const float* __restrict__ in,
                                                   __bf16* __restrict__ out,
                                                   size_t n) {
    size_t i = (size_t)blockIdx.x * blockDim.x + threadIdx.x;
    if (i < n) out[i] = (__bf16)in[i];
}

// =======================================================================
// Tiled bf16 WMMA GEMM: 128x128 block tile, BK=64 (two K=32 WMMA steps),
// double-buffered LDS fed by the Tensor Data Mover (2 TDM ops per K-step,
// issued by waves 0/1; all 8 waves spend their slots on WMMA).
// MODE 0: QKV epilogue -> scatter q/k/v to [B,H,T,D] bf16 (+bias)
// MODE 1: out-proj epilogue -> fp32 [M,N] (+bias)
// =======================================================================
template <int MODE>
__global__ __launch_bounds__(256) void gemm_bf16_kernel(
    const __bf16* __restrict__ A,   // [M,K] row-major bf16
    const __bf16* __restrict__ Bw,  // [K,N] row-major bf16
    const float*  __restrict__ bias,
    int N, int K,
    float*  __restrict__ outF,
    __bf16* __restrict__ qO, __bf16* __restrict__ kO, __bf16* __restrict__ vO) {

    constexpr int BM = 128, BN = 128, BK = 64;
    __shared__ __align__(16) __bf16 As[2][BM * BK];   // 2 x 16KB
    __shared__ __align__(16) __bf16 Bs[2][BK * BN];   // 2 x 16KB

    const int tid  = threadIdx.x;
    const int wave = tid >> 5;
    const int lane = tid & 31;
    const int wm = wave >> 2;       // 0..1
    const int wn = wave & 3;        // 0..3
    const int bm = blockIdx.y * BM;
    const int bn = blockIdx.x * BN;

    // stage one A/B tile pair into LDS buffer `bufi` via the TDM (scalar branch:
    // TDM ignores EXEC, so only the issuing wave may reach the instruction)
    auto stage = [&](int k0, int bufi) {
        int w = __builtin_amdgcn_readfirstlane(wave);
        if (w == 0) {
            // A tile: BM rows x BK cols, row stride K  -> As[row*BK + k]
            tdm_load_2d(lds_off_of(&As[bufi][0]),
                        A + (size_t)bm * K + k0,
                        (unsigned)BK, (unsigned)BM,   // tensor dims (== tile, in-bounds)
                        (unsigned)BK, (unsigned)BM,   // tile dims
                        (unsigned)K);
        } else if (w == 1) {
            // B tile: BK rows x BN cols, row stride N  -> Bs[k*BN + n]
            tdm_load_2d(lds_off_of(&Bs[bufi][0]),
                        Bw + (size_t)k0 * N + bn,
                        (unsigned)BN, (unsigned)BK,
                        (unsigned)BN, (unsigned)BK,
                        (unsigned)N);
        }
    };

    v8f acc[4][2] = {};
    const int nk = K / BK;

    stage(0, 0);
    wait_tensor0();
    __syncthreads();

    int buf = 0;
    for (int ks = 0; ks < nk; ++ks) {
        if (ks + 1 < nk) stage((ks + 1) * BK, buf ^ 1);   // overlap next tile DMA

        // ---- compute on current buffer: 2 x (K=32) sub-steps, 8 WMMA each ----
        #pragma unroll
        for (int kc = 0; kc < 2; ++kc) {
            v16bf afr[4], bfr[2];
            const int arow = lane & 15;
            const int akb  = kc * 32 + (lane >> 4) * 16;     // A: K base
            #pragma unroll
            for (int mt = 0; mt < 4; ++mt) {
                int row = wm * 64 + mt * 16 + arow;
                #pragma unroll
                for (int i = 0; i < 16; ++i) {
                    int kk = akb + i;
                    afr[mt][i] = As[buf][row * BK + kk];
                }
            }
            const int bcol = lane & 15;
            const int bkb  = kc * 32 + (lane >> 4) * 16;     // B: K base
            #pragma unroll
            for (int nt = 0; nt < 2; ++nt) {
                int col = wn * 32 + nt * 16 + bcol;
                #pragma unroll
                for (int i = 0; i < 16; ++i)
                    bfr[nt][i] = Bs[buf][(bkb + i) * BN + col];
            }
            #pragma unroll
            for (int mt = 0; mt < 4; ++mt)
                #pragma unroll
                for (int nt = 0; nt < 2; ++nt)
                    acc[mt][nt] = __builtin_amdgcn_wmma_f32_16x16x32_bf16(
                        false, afr[mt], false, bfr[nt], (short)0, acc[mt][nt],
                        false, false);
        }

        wait_tensor0();      // issuing waves drain their TDM; others pass
        __syncthreads();     // visible to all waves; safe to flip buffers
        buf ^= 1;
    }

    // ---- epilogue: C/D layout n = lane&15, m = r + (lane>>4)*8 ----
    #pragma unroll
    for (int mt = 0; mt < 4; ++mt) {
        #pragma unroll
        for (int nt = 0; nt < 2; ++nt) {
            #pragma unroll
            for (int r = 0; r < 8; ++r) {
                int m = bm + wm * 64 + mt * 16 + r + (lane >> 4) * 8;
                int n = bn + wn * 32 + nt * 16 + (lane & 15);
                float val = acc[mt][nt][r] + bias[n];
                if constexpr (MODE == 0) {
                    int part = n / C_, cc = n % C_;
                    int h = cc >> 6, d = cc & 63;
                    int b = m / T_, t = m % T_;
                    size_t idx = (((size_t)b * H_ + h) * T_ + t) * D_ + d;
                    __bf16 bv = (__bf16)val;
                    if (part == 0)      qO[idx] = bv;
                    else if (part == 1) kO[idx] = bv;
                    else                vO[idx] = bv;
                } else {
                    outF[(size_t)m * N + n] = val;
                }
            }
        }
    }
}

// =======================================================================
// Flash attention: block = 128 queries of one (b,h); 8 waves, each wave
// owns 16 query rows. 32-key K/V tiles double-buffered in LDS, staged by
// the TDM (waves 0/1); Q staged via async global->LDS. Online softmax;
// scores never touch HBM.
// =======================================================================
__global__ __launch_bounds__(256) void flash_attn_kernel(
    const __bf16* __restrict__ Q,   // [B,H,T,D]
    const __bf16* __restrict__ Km,  // [B,H,T,D]
    const __bf16* __restrict__ Vm,  // [B,H,T,D]
    __bf16* __restrict__ O) {       // [B*T, C] bf16

    constexpr int QT = 128;
    const int bh = blockIdx.y;                 // 0..B*H-1
    const int qb = blockIdx.x * QT;
    const __bf16* Qp = Q  + (size_t)bh * T_ * D_;
    const __bf16* Kp = Km + (size_t)bh * T_ * D_;
    const __bf16* Vp = Vm + (size_t)bh * T_ * D_;

    __shared__ __align__(16) __bf16 Qs[QT * D_];        // 16KB
    __shared__ __align__(16) __bf16 Ks[2][32 * D_];     // 2 x 4KB
    __shared__ __align__(16) __bf16 Vs[2][32 * D_];     // 2 x 4KB
    __shared__ __align__(16) __bf16 Ps[8][16 * 32];     // 8KB, per-wave P tile

    const int tid  = threadIdx.x;
    const int wave = tid >> 5;
    const int lane = tid & 31;

    // TDM-stage one 32x64 K tile and one V tile (waves 0/1, scalar branch)
    auto stage_kv = [&](int kv, int bufi) {
        int w = __builtin_amdgcn_readfirstlane(wave);
        if (w == 0) {
            tdm_load_2d(lds_off_of(&Ks[bufi][0]), Kp + (size_t)kv * D_,
                        64u, 32u, 64u, 32u, 64u);
        } else if (w == 1) {
            tdm_load_2d(lds_off_of(&Vs[bufi][0]), Vp + (size_t)kv * D_,
                        64u, 32u, 64u, 32u, 64u);
        }
    };

    // ---- stage Q tile (128x64) via async copies + first K/V tile via TDM ----
    #pragma unroll
    for (int c = tid; c < QT * D_ / 8; c += 256) {      // 1024 chunks, 4/thread
        int row = c >> 3, col8 = (c & 7) * 8;
        async_copy_b128(&Qs[row * D_ + col8],
                        Qp + (size_t)(qb + row) * D_ + col8);
    }
    stage_kv(0, 0);
    wait_async0();
    wait_tensor0();
    __syncthreads();

    // ---- per-wave Q fragments for the two K=32 chunks of D=64 ----
    v16bf qf[2];
    {
        int row = wave * 16 + (lane & 15);
        int akb = (lane >> 4) * 16;
        #pragma unroll
        for (int c2 = 0; c2 < 2; ++c2)
            #pragma unroll
            for (int i = 0; i < 16; ++i) {
                int kk = c2 * 32 + akb + i;
                qf[c2][i] = Qs[row * D_ + kk];
            }
    }

    float mi[8], li[8];
    #pragma unroll
    for (int r = 0; r < 8; ++r) { mi[r] = -1e30f; li[r] = 0.0f; }
    v8f oacc[4] = {};
    const float scale = 0.125f;  // 1/sqrt(64)

    const int nkv = (qb + QT < T_) ? (qb + QT) : T_;
    int buf = 0;
    for (int kv = 0; kv < nkv; kv += 32) {
        if (kv + 32 < nkv) stage_kv(kv + 32, buf ^ 1);  // overlap next tile DMA

        // ---- S = Q @ K^T : 2 key-chunks x 2 d-chunks = 4 WMMAs ----
        v8f s[2] = {};
        #pragma unroll
        for (int nc = 0; nc < 2; ++nc) {
            int key = nc * 16 + (lane & 15);
            #pragma unroll
            for (int c2 = 0; c2 < 2; ++c2) {
                v16bf bf;
                int db = (lane >> 4) * 16 + c2 * 32;
                #pragma unroll
                for (int i = 0; i < 16; ++i)
                    bf[i] = Ks[buf][key * D_ + db + i];
                s[nc] = __builtin_amdgcn_wmma_f32_16x16x32_bf16(
                    false, qf[c2], false, bf, (short)0, s[nc], false, false);
            }
        }

        // ---- causal mask + scale ----
        const int qrow0 = qb + wave * 16 + (lane >> 4) * 8;
        #pragma unroll
        for (int nc = 0; nc < 2; ++nc) {
            int key = kv + nc * 16 + (lane & 15);
            #pragma unroll
            for (int r = 0; r < 8; ++r) {
                float v = s[nc][r] * scale;
                s[nc][r] = (key <= qrow0 + r) ? v : -1e30f;
            }
        }

        // ---- online softmax: row reductions across 16 lanes ----
        float corr[8];
        #pragma unroll
        for (int r = 0; r < 8; ++r) {
            float v = fmaxf(s[0][r], s[1][r]);
            #pragma unroll
            for (int off = 8; off >= 1; off >>= 1)
                v = fmaxf(v, __shfl_xor(v, off, 16));
            float mnew = fmaxf(mi[r], v);
            corr[r] = __expf(mi[r] - mnew);
            mi[r] = mnew;
        }
        #pragma unroll
        for (int nc = 0; nc < 2; ++nc)
            #pragma unroll
            for (int r = 0; r < 8; ++r)
                s[nc][r] = __expf(s[nc][r] - mi[r]);
        #pragma unroll
        for (int r = 0; r < 8; ++r) {
            float v = s[0][r] + s[1][r];
            #pragma unroll
            for (int off = 8; off >= 1; off >>= 1)
                v += __shfl_xor(v, off, 16);
            li[r] = li[r] * corr[r] + v;
        }
        #pragma unroll
        for (int dc = 0; dc < 4; ++dc)
            #pragma unroll
            for (int r = 0; r < 8; ++r)
                oacc[dc][r] *= corr[r];

        // ---- restage P through per-wave LDS: C-layout f32 -> A-layout bf16 ----
        #pragma unroll
        for (int nc = 0; nc < 2; ++nc)
            #pragma unroll
            for (int r = 0; r < 8; ++r) {
                int m = r + (lane >> 4) * 8;
                int n = nc * 16 + (lane & 15);
                Ps[wave][m * 32 + n] = (__bf16)s[nc][r];
            }
        asm volatile("s_wait_dscnt 0" ::: "memory");  // order ds_write -> ds_read (same wave)
        v16bf pf;
        {
            int row = lane & 15;
            int akb = (lane >> 4) * 16;
            #pragma unroll
            for (int i = 0; i < 16; ++i) {
                int kk = akb + i;
                pf[i] = Ps[wave][row * 32 + kk];
            }
        }

        // ---- O += P @ V : 4 d-chunks = 4 WMMAs ----
        #pragma unroll
        for (int dc = 0; dc < 4; ++dc) {
            v16bf vf;
            int dcol = dc * 16 + (lane & 15);
            int kb2  = (lane >> 4) * 16;
            #pragma unroll
            for (int i = 0; i < 16; ++i)
                vf[i] = Vs[buf][(kb2 + i) * D_ + dcol];
            oacc[dc] = __builtin_amdgcn_wmma_f32_16x16x32_bf16(
                false, pf, false, vf, (short)0, oacc[dc], false, false);
        }

        wait_tensor0();      // next K/V tile landed (issuing waves)
        __syncthreads();     // all waves done reading old buffer
        buf ^= 1;
    }

    // ---- write O back to [B*T, C] bf16 (transpose heads back) ----
    const int b = bh / H_, h = bh % H_;
    #pragma unroll
    for (int dc = 0; dc < 4; ++dc) {
        #pragma unroll
        for (int r = 0; r < 8; ++r) {
            int t = qb + wave * 16 + r + (lane >> 4) * 8;
            int d = dc * 16 + (lane & 15);
            float inv = (li[r] > 0.0f) ? 1.0f / li[r] : 0.0f;
            O[((size_t)b * T_ + t) * C_ + h * D_ + d] = (__bf16)(oacc[dc][r] * inv);
        }
    }
}

// =======================================================================
// Host launch
// =======================================================================
extern "C" void kernel_launch(void* const* d_in, const int* in_sizes, int n_in,
                              void* d_out, int out_size, void* d_ws, size_t ws_size,
                              hipStream_t stream) {
    (void)in_sizes; (void)n_in; (void)out_size; (void)ws_size;

    const float* x      = (const float*)d_in[0];
    const float* qkv_w  = (const float*)d_in[1];
    const float* qkv_b  = (const float*)d_in[2];
    const float* out_w  = (const float*)d_in[3];
    const float* out_b  = (const float*)d_in[4];
    float* out = (float*)d_out;

    // workspace carve-out (all sizes 16B-aligned)
    char* ws = (char*)d_ws;
    __bf16* xbf   = (__bf16*)ws; ws += (size_t)BT_ * C_ * 2;
    __bf16* wqkv  = (__bf16*)ws; ws += (size_t)C_ * C3_ * 2;
    __bf16* wout  = (__bf16*)ws; ws += (size_t)C_ * C_ * 2;
    __bf16* qbf   = (__bf16*)ws; ws += (size_t)B_ * H_ * T_ * D_ * 2;
    __bf16* kbf   = (__bf16*)ws; ws += (size_t)B_ * H_ * T_ * D_ * 2;
    __bf16* vbf   = (__bf16*)ws; ws += (size_t)B_ * H_ * T_ * D_ * 2;
    __bf16* attbf = (__bf16*)ws; ws += (size_t)BT_ * C_ * 2;

    // 1) fp32 -> bf16 conversions
    {
        size_t n;
        n = (size_t)BT_ * C_;
        f2bf_kernel<<<(unsigned)((n + 255) / 256), 256, 0, stream>>>(x, xbf, n);
        n = (size_t)C_ * C3_;
        f2bf_kernel<<<(unsigned)((n + 255) / 256), 256, 0, stream>>>(qkv_w, wqkv, n);
        n = (size_t)C_ * C_;
        f2bf_kernel<<<(unsigned)((n + 255) / 256), 256, 0, stream>>>(out_w, wout, n);
    }

    // 2) QKV projection: [8192,1024] x [1024,3072] -> q/k/v [B,H,T,D]
    {
        dim3 grid(C3_ / 128, BT_ / 128);   // (24, 64)
        gemm_bf16_kernel<0><<<grid, 256, 0, stream>>>(
            xbf, wqkv, qkv_b, C3_, C_, nullptr, qbf, kbf, vbf);
    }

    // 3) Flash attention -> [B*T, C] bf16
    {
        dim3 grid(T_ / 128, B_ * H_);      // (16, 64)
        flash_attn_kernel<<<grid, 256, 0, stream>>>(qbf, kbf, vbf, attbf);
    }

    // 4) Output projection: [8192,1024] x [1024,1024] + bias -> fp32 d_out
    {
        dim3 grid(C_ / 128, BT_ / 128);    // (8, 64)
        gemm_bf16_kernel<1><<<grid, 256, 0, stream>>>(
            attbf, wout, out_b, C_, C_, out, nullptr, nullptr, nullptr);
    }
}
